// LIE_44126493999142
// MI455X (gfx1250) — matrix operand order repaired and checked by reading, not compiled, not run on hardware
//
#include <hip/hip_runtime.h>
#include <hip/hip_bf16.h>
#include <math.h>

// ---------------------------------------------------------------------------
// Problem constants (match reference): B=8, T=2048, D=1024, W=64
// ---------------------------------------------------------------------------
#define BB 8
#define TT 2048
#define DD 1024
#define WW 64
#define LN_EPS 1e-5f

typedef __attribute__((ext_vector_type(16))) __bf16 v16bf;
typedef __attribute__((ext_vector_type(8)))  __bf16 v8bf;
typedef __attribute__((ext_vector_type(8)))  float  v8f;

__device__ __forceinline__ __bf16 f2bf(float x) {
    unsigned u = __builtin_bit_cast(unsigned, x);
    unsigned r = (u + 0x7FFFu + ((u >> 16) & 1u)) >> 16;   // round-to-nearest-even
    unsigned short s = (unsigned short)r;
    return __builtin_bit_cast(__bf16, s);
}
__device__ __forceinline__ float bf2f(__bf16 h) {
    unsigned short s = __builtin_bit_cast(unsigned short, h);
    unsigned u = ((unsigned)s) << 16;
    return __builtin_bit_cast(float, u);
}

// ---------------------------------------------------------------------------
// Kernel 1: q = feature @ wq_w + wq_b   (M=B*T=16384, N=K=1024)
// Split-bf16 WMMA: (a_hi+a_lo)(b_hi+b_lo) ~ hi*hi + hi*lo + lo*hi, f32 accum.
// Block tile 128(M) x 64(N), 8 waves (4x2), each wave a 32x32 (2x2 WMMA tiles)
// -> 12 v_wmma per k-step per wave.
// ---------------------------------------------------------------------------
#define LDA 40   // padded LDS row stride (elements): 80B keeps 16B alignment,
                 // 20-word stride -> conflict-free b128 fragment loads

__global__ __launch_bounds__(256)
void qgemm_kernel(const float* __restrict__ A, const float* __restrict__ Bw,
                  const float* __restrict__ bias, float* __restrict__ Q) {
    __shared__ __align__(16) __bf16 Ah[128][LDA];
    __shared__ __align__(16) __bf16 Al[128][LDA];
    __shared__ __align__(16) __bf16 Bth[64][LDA];   // transposed: [n][k]
    __shared__ __align__(16) __bf16 Btl[64][LDA];

    const int tid  = threadIdx.x;
    const int m0   = blockIdx.y * 128;
    const int n0   = blockIdx.x * 64;
    const int wave = tid >> 5;
    const int lane = tid & 31;
    const int wm   = wave & 3;       // 4 wave-rows (M), 32 rows each
    const int wn   = wave >> 2;      // 2 wave-cols (N), 32 cols each
    const int h    = lane >> 4;      // lane half
    const int ln   = lane & 15;

    v8f acc[2][2];
    #pragma unroll
    for (int mi = 0; mi < 2; ++mi)
        #pragma unroll
        for (int ni = 0; ni < 2; ++ni)
            acc[mi][ni] = (v8f){0.f,0.f,0.f,0.f,0.f,0.f,0.f,0.f};

    for (int k0 = 0; k0 < DD; k0 += 32) {
        // ---- stage A tile 128x32 (four float4 per thread), split hi/lo bf16
        #pragma unroll
        for (int it = 0; it < 4; ++it) {
            int idx = tid + it * 256;
            int row = idx >> 3;
            int c4  = (idx & 7) << 2;
            const float4 v = *(const float4*)&A[(size_t)(m0 + row) * DD + k0 + c4];
            float vv[4] = {v.x, v.y, v.z, v.w};
            #pragma unroll
            for (int j = 0; j < 4; ++j) {
                __bf16 hi = f2bf(vv[j]);
                __bf16 lo = f2bf(vv[j] - bf2f(hi));
                Ah[row][c4 + j] = hi;
                Al[row][c4 + j] = lo;
            }
        }
        // ---- stage B tile 32(k)x64(n), transposed into [n][k]
        #pragma unroll
        for (int it = 0; it < 2; ++it) {
            int idx = tid + it * 256;
            int kr  = idx >> 4;              // 16 float4 per k-row of 64 cols
            int c4  = (idx & 15) << 2;
            const float4 v = *(const float4*)&Bw[(size_t)(k0 + kr) * DD + n0 + c4];
            float vv[4] = {v.x, v.y, v.z, v.w};
            #pragma unroll
            for (int j = 0; j < 4; ++j) {
                __bf16 hi = f2bf(vv[j]);
                __bf16 lo = f2bf(vv[j] - bf2f(hi));
                Bth[c4 + j][kr] = hi;
                Btl[c4 + j][kr] = lo;
            }
        }
        __syncthreads();

        // ---- fragments per the 16-bit A 16x32 / B 32x16 VGPR layouts
        v16bf afh[2], afl[2], bfh[2], bfl[2];
        #pragma unroll
        for (int mi = 0; mi < 2; ++mi) {
            const int r = wm * 32 + mi * 16 + ln;
            v8bf c0h = *(const v8bf*)&Ah[r][h * 8];
            v8bf c1h = *(const v8bf*)&Ah[r][16 + h * 8];
            v8bf c0l = *(const v8bf*)&Al[r][h * 8];
            v8bf c1l = *(const v8bf*)&Al[r][16 + h * 8];
            afh[mi] = __builtin_shufflevector(c0h, c1h, 0,1,2,3,4,5,6,7,8,9,10,11,12,13,14,15);
            afl[mi] = __builtin_shufflevector(c0l, c1l, 0,1,2,3,4,5,6,7,8,9,10,11,12,13,14,15);
        }
        #pragma unroll
        for (int ni = 0; ni < 2; ++ni) {
            const int nn = wn * 32 + ni * 16 + ln;
            v8bf c0h = *(const v8bf*)&Bth[nn][h * 8];
            v8bf c1h = *(const v8bf*)&Bth[nn][16 + h * 8];
            v8bf c0l = *(const v8bf*)&Btl[nn][h * 8];
            v8bf c1l = *(const v8bf*)&Btl[nn][16 + h * 8];
            bfh[ni] = __builtin_shufflevector(c0h, c1h, 0,1,2,3,4,5,6,7,8,9,10,11,12,13,14,15);
            bfl[ni] = __builtin_shufflevector(c0l, c1l, 0,1,2,3,4,5,6,7,8,9,10,11,12,13,14,15);
        }

        #pragma unroll
        for (int mi = 0; mi < 2; ++mi)
            #pragma unroll
            for (int ni = 0; ni < 2; ++ni) {
                acc[mi][ni] = __builtin_amdgcn_wmma_f32_16x16x32_bf16(false, afh[mi], false, bfh[ni], (short)0, acc[mi][ni], false, false);
                acc[mi][ni] = __builtin_amdgcn_wmma_f32_16x16x32_bf16(false, afh[mi], false, bfl[ni], (short)0, acc[mi][ni], false, false);
                acc[mi][ni] = __builtin_amdgcn_wmma_f32_16x16x32_bf16(false, afl[mi], false, bfh[ni], (short)0, acc[mi][ni], false, false);
            }
        __syncthreads();
    }

    #pragma unroll
    for (int ni = 0; ni < 2; ++ni) {
        const int nglob = n0 + wn * 32 + ni * 16 + ln;
        const float bn = bias[nglob];
        #pragma unroll
        for (int mi = 0; mi < 2; ++mi) {
            #pragma unroll
            for (int i = 0; i < 8; ++i) {
                int m = m0 + wm * 32 + mi * 16 + h * 8 + i;  // VGPR i -> M=i+8*half
                Q[(size_t)m * DD + nglob] = acc[mi][ni][i] + bn;
            }
        }
    }
}

// ---------------------------------------------------------------------------
// Kernel 2: sequential windowed-attention scan. One workgroup per batch.
// 512 threads = 16 waves; each wave owns 4 window rows IN REGISTERS
// (4x32 f32/lane). Ring buffer is permutation-invariant under softmax.
// LDS holds only qw (4KB), 16 partial rows (64KB), scores/alpha, vtmp.
// The q row for step t+1 is loaded from HBM during step t (latency hiding).
// Overwrites q rows with v rows (workspace reuse).
// ---------------------------------------------------------------------------
__global__ __launch_bounds__(512)
void scan_kernel(const float* __restrict__ feat, const float* __restrict__ w2w,
                 const float* __restrict__ w2b, float* __restrict__ qv) {
    __shared__ float qwl[DD];            // qw[d] = q[d]*w2_w[d]
    __shared__ float part[16][DD];       // per-wave alpha-weighted partial rows
    __shared__ float vtmp[DD];           // broadcast buffer for the new v row
    __shared__ float scores[WW];
    __shared__ float alpha[WW];

    const int b    = blockIdx.x;
    const int tid  = threadIdx.x;
    const int wave = tid >> 5;           // 0..15
    const int lane = tid & 31;

    const float wd0 = w2w[tid];
    const float wd1 = w2w[tid + 512];
    const float zscore = w2b[0];         // score of the appended all-zeros row

    // ---- init: wave w owns window rows 4w..4w+3 in registers
    float r0[32], r1[32], r2[32], r3[32];
    {
        const size_t base = ((size_t)b * TT + (size_t)(4 * wave)) * DD;
        #pragma unroll
        for (int e = 0; e < 32; ++e) {
            const int d = e * 32 + lane;
            r0[e] = feat[base + 0 * DD + d];
            r1[e] = feat[base + 1 * DD + d];
            r2[e] = feat[base + 2 * DD + d];
            r3[e] = feat[base + 3 * DD + d];
        }
    }

    // preload q row for t = W
    float qa = qv[((size_t)b * TT + WW) * DD + tid];
    float qb = qv[((size_t)b * TT + WW) * DD + tid + 512];

    for (int t = WW; t < TT; ++t) {
        const size_t rowoff = ((size_t)b * TT + t) * DD;
        // publish qw for this step
        qwl[tid]       = qa * wd0;
        qwl[tid + 512] = qb * wd1;
        // kick off next step's q row load (consumed next iteration)
        const int tn = (t + 1 < TT) ? (t + 1) : t;
        const size_t nrow = ((size_t)b * TT + tn) * DD;
        const float na = qv[nrow + tid];
        const float nb = qv[nrow + tid + 512];
        __syncthreads();                                   // (1)

        // ---- scores from registers: s_j = dot(row_{4w+j}, qw)
        float s0 = 0.f, s1 = 0.f, s2 = 0.f, s3 = 0.f;
        #pragma unroll
        for (int e = 0; e < 32; ++e) {
            const float qq = qwl[e * 32 + lane];
            s0 = fmaf(r0[e], qq, s0);
            s1 = fmaf(r1[e], qq, s1);
            s2 = fmaf(r2[e], qq, s2);
            s3 = fmaf(r3[e], qq, s3);
        }
        #pragma unroll
        for (int off = 16; off > 0; off >>= 1) {
            s0 += __shfl_xor(s0, off, 32);
            s1 += __shfl_xor(s1, off, 32);
            s2 += __shfl_xor(s2, off, 32);
            s3 += __shfl_xor(s3, off, 32);
        }
        if (lane == 0) {
            scores[4 * wave + 0] = s0;
            scores[4 * wave + 1] = s1;
            scores[4 * wave + 2] = s2;
            scores[4 * wave + 3] = s3;
        }
        __syncthreads();                                   // (2)

        // ---- softmax over 65 scores (64 rows + zero row) by wave 0
        if (wave == 0) {
            const float sa = scores[lane];
            const float sc = scores[lane + 32];
            float m = fmaxf(sa, sc);
            #pragma unroll
            for (int off = 16; off > 0; off >>= 1)
                m = fmaxf(m, __shfl_xor(m, off, 32));
            m = fmaxf(m, zscore);
            const float ea = __expf(sa - m);
            const float ec = __expf(sc - m);
            float sum = ea + ec;
            #pragma unroll
            for (int off = 16; off > 0; off >>= 1)
                sum += __shfl_xor(sum, off, 32);
            sum += __expf(zscore - m);      // zero row joins denominator only
            const float inv = 1.f / sum;
            alpha[lane]      = ea * inv;
            alpha[lane + 32] = ec * inv;
        }
        __syncthreads();                                   // (3)

        // ---- per-wave alpha-weighted partial row
        const float a0 = alpha[4 * wave + 0];
        const float a1 = alpha[4 * wave + 1];
        const float a2 = alpha[4 * wave + 2];
        const float a3 = alpha[4 * wave + 3];
        #pragma unroll
        for (int e = 0; e < 32; ++e) {
            part[wave][e * 32 + lane] =
                fmaf(a0, r0[e], fmaf(a1, r1[e], fmaf(a2, r2[e], a3 * r3[e])));
        }
        __syncthreads();                                   // (4)

        // ---- flat reduction of 16 partials -> v row; publish + store
        float acc0 = 0.f, acc1 = 0.f;
        #pragma unroll
        for (int i = 0; i < 16; ++i) {
            acc0 += part[i][tid];
            acc1 += part[i][tid + 512];
        }
        qv[rowoff + tid]       = acc0;   // overwrite q with v (ws reuse)
        qv[rowoff + tid + 512] = acc1;
        vtmp[tid]       = acc0;
        vtmp[tid + 512] = acc1;
        __syncthreads();                                   // (5)

        // ---- ring-buffer insert: owner wave pulls new row into registers.
        // Compile-time register indices via switch (no dynamic indexing).
        const int slot = t & (WW - 1);
        if (wave == (slot >> 2)) {
            switch (slot & 3) {
            case 0: {
                #pragma unroll
                for (int e = 0; e < 32; ++e) r0[e] = vtmp[e * 32 + lane];
                break;
            }
            case 1: {
                #pragma unroll
                for (int e = 0; e < 32; ++e) r1[e] = vtmp[e * 32 + lane];
                break;
            }
            case 2: {
                #pragma unroll
                for (int e = 0; e < 32; ++e) r2[e] = vtmp[e * 32 + lane];
                break;
            }
            default: {
                #pragma unroll
                for (int e = 0; e < 32; ++e) r3[e] = vtmp[e * 32 + lane];
                break;
            }
            }
        }
        // vtmp is rewritten only after 4 barriers next step; qwl (written at
        // the next loop top) is a different buffer -> no extra barrier needed.
        qa = na; qb = nb;
    }
}

// ---------------------------------------------------------------------------
// Kernel 3: y = tanh(v_local)*feature + feature; LayerNorm over D.
// One block (256 threads) per row; v = feature for t < W, else scan output.
// ---------------------------------------------------------------------------
__global__ __launch_bounds__(256)
void ln_kernel(const float* __restrict__ feat, const float* __restrict__ vv,
               const float* __restrict__ g, const float* __restrict__ be,
               float* __restrict__ out) {
    const int row = blockIdx.x;          // b*T + t
    const int t   = row & (TT - 1);
    const int tid = threadIdx.x;
    const int wave = tid >> 5;
    const int lane = tid & 31;
    const size_t off = (size_t)row * DD;

    __shared__ float redS[8];
    __shared__ float redQ[8];

    float y[4];
    float sum = 0.f, sq = 0.f;
    #pragma unroll
    for (int j = 0; j < 4; ++j) {
        const int d = tid + j * 256;
        const float f = feat[off + d];
        const float v = (t < WW) ? f : vv[off + d];
        const float yy = tanhf(v) * f + f;
        y[j] = yy;
        sum += yy;
        sq  += yy * yy;
    }
    #pragma unroll
    for (int o = 16; o > 0; o >>= 1) {
        sum += __shfl_xor(sum, o, 32);
        sq  += __shfl_xor(sq,  o, 32);
    }
    if (lane == 0) { redS[wave] = sum; redQ[wave] = sq; }
    __syncthreads();
    float tsum = 0.f, tsq = 0.f;
    #pragma unroll
    for (int i = 0; i < 8; ++i) { tsum += redS[i]; tsq += redQ[i]; }

    const float mu   = tsum * (1.f / DD);
    const float var  = tsq * (1.f / DD) - mu * mu;
    const float rstd = rsqrtf(var + LN_EPS);
    #pragma unroll
    for (int j = 0; j < 4; ++j) {
        const int d = tid + j * 256;
        out[off + d] = (y[j] - mu) * rstd * g[d] + be[d];
    }
}

// ---------------------------------------------------------------------------
extern "C" void kernel_launch(void* const* d_in, const int* in_sizes, int n_in,
                              void* d_out, int out_size, void* d_ws, size_t ws_size,
                              hipStream_t stream) {
    const float* feature = (const float*)d_in[0];
    const float* wq_w    = (const float*)d_in[1];
    const float* wq_b    = (const float*)d_in[2];
    const float* w2_w    = (const float*)d_in[3];
    const float* w2_b    = (const float*)d_in[4];
    const float* ln_g    = (const float*)d_in[5];
    const float* ln_b    = (const float*)d_in[6];
    float* out = (float*)d_out;
    float* q   = (float*)d_ws;      // B*T*D f32 = 64 MB; q then overwritten by v

    dim3 g1(DD / 64, (BB * TT) / 128);
    qgemm_kernel<<<g1, 256, 0, stream>>>(feature, wq_w, wq_b, q);
    scan_kernel<<<BB, 512, 0, stream>>>(feature, w2_w, w2_b, q);
    ln_kernel<<<BB * TT, 256, 0, stream>>>(feature, q, ln_g, ln_b, out);
}
